// SingleHeadModel_27642409517469
// MI455X (gfx1250) — matrix operand order repaired and checked by reading, not compiled, not run on hardware
//
#include <hip/hip_runtime.h>

#define B_ 2
#define T_ 2048
#define D_ 1024
#define V_ 32000

typedef __attribute__((ext_vector_type(16))) __bf16 v16bf;
typedef __attribute__((ext_vector_type(8)))  float  v8f;

static __device__ __forceinline__ unsigned short f2bf(float f) {
  union { float f; unsigned u; } v; v.f = f;
  unsigned r = v.u + 0x7FFFu + ((v.u >> 16) & 1u);   // round-to-nearest-even
  return (unsigned short)(r >> 16);
}

static __device__ __forceinline__ void wait_async0() {
#if __has_builtin(__builtin_amdgcn_s_wait_asynccnt)
  __builtin_amdgcn_s_wait_asynccnt(0);
#else
  asm volatile("s_wait_asynccnt 0x0" ::: "memory");
#endif
}
static __device__ __forceinline__ void wait_async6() {
#if __has_builtin(__builtin_amdgcn_s_wait_asynccnt)
  __builtin_amdgcn_s_wait_asynccnt(6);
#else
  asm volatile("s_wait_asynccnt 0x6" ::: "memory");
#endif
}

static __device__ __forceinline__ void async_copy16(const unsigned short* gsrc,
                                                    void* ldst) {
  unsigned lds = (unsigned)(size_t)ldst;   // low 32 bits of generic ptr = LDS offset
  asm volatile("global_load_async_to_lds_b128 %0, %1, off"
               :: "v"(lds), "v"(gsrc) : "memory");
}

// ---------------------------------------------------------------------------
// x = embed[inputs] + pos_embed  ->  bf16
// ---------------------------------------------------------------------------
__global__ __launch_bounds__(256)
void embed_kernel(const int* __restrict__ inputs, const float* __restrict__ embed,
                  const float* __restrict__ pos, unsigned short* __restrict__ xbf) {
  int row = blockIdx.x;                    // 0 .. B*T-1
  int t = row % T_;
  int tok = inputs[row];
  const float* e = embed + (size_t)tok * D_;
  const float* p = pos + (size_t)t * D_;
  unsigned short* o = xbf + (size_t)row * D_;
  for (int d = threadIdx.x; d < D_; d += 256)
    o[d] = f2bf(e[d] + p[d]);
}

__global__ __launch_bounds__(256)
void f32_to_bf16_kernel(const float* __restrict__ in, unsigned short* __restrict__ out,
                        long n) {
  long i = (long)blockIdx.x * 256 + threadIdx.x;
  if (i < n) out[i] = f2bf(in[i]);
}

// ---------------------------------------------------------------------------
// out[c][r] = bf16(in[r][c])  -- LDS-tiled transposing convert (32x32 tiles)
// ---------------------------------------------------------------------------
__global__ __launch_bounds__(256)
void transpose_f32_to_bf16_kernel(const float* __restrict__ in,
                                  unsigned short* __restrict__ out,
                                  int rows, int cols, long strideIn, long strideOut) {
  __shared__ float tile[32][33];
  in  += strideIn  * blockIdx.z;
  out += strideOut * blockIdx.z;
  int c0 = blockIdx.x * 32, r0 = blockIdx.y * 32;
  int tx = threadIdx.x & 31, ty = threadIdx.x >> 5;     // 8 rows per pass
  #pragma unroll
  for (int i = 0; i < 32; i += 8)
    tile[ty + i][tx] = in[(size_t)(r0 + ty + i) * cols + c0 + tx];
  __syncthreads();
  #pragma unroll
  for (int i = 0; i < 32; i += 8)
    out[(size_t)(c0 + ty + i) * rows + r0 + tx] = f2bf(tile[tx][ty + i]);
}

// ---------------------------------------------------------------------------
// bf16 WMMA GEMM (TN): C(f32, MxN) = A(bf16 MxK row) * Bt(bf16 NxK row)^T
// Block = 256 threads (8 waves), tile 256(M) x 128(N), K-step 32.
// Wave grid 4(M) x 2(N); each wave: 4x4 fragments of 16x16 => 16 WMMAs/K-step.
// Double-buffered LDS; tile k+1 prefetched via GLOBAL_LOAD_ASYNC_TO_LDS_B128
// while tile k feeds the WMMAs; partial wait s_wait_asynccnt 6 (in-order done).
// ---------------------------------------------------------------------------
__global__ __launch_bounds__(256)
void gemm_bf16_tn_kernel(const unsigned short* __restrict__ A,
                         const unsigned short* __restrict__ Bt,
                         float* __restrict__ C,
                         int Kd, int lda, int ldb, int ldc,
                         long strideA, long strideB, long strideC)
{
  constexpr int BM = 256, BN = 128, BK = 32, LDSW = BK + 8;  // row = 80B (16B aligned)
  __shared__ __align__(16) unsigned short As[2][BM][LDSW];
  __shared__ __align__(16) unsigned short Bs[2][BN][LDSW];

  const int bz = blockIdx.z;
  A  += strideA * bz;
  Bt += strideB * bz;
  C  += strideC * bz;

  const int n0 = blockIdx.x * BN;
  const int m0 = blockIdx.y * BM;

  const int tid  = threadIdx.x;
  const int lane = tid & 31;
  const int w    = tid >> 5;
  const int half = lane >> 4;
  const int l16  = lane & 15;
  const int waveM = w & 3;     // 64 rows per wave
  const int waveN = w >> 2;    // 64 cols per wave

  // per-thread staging coordinates (4 A chunks + 2 B chunks, 16B each)
  const int arow0 = tid >> 2, acj = tid & 3;      // + 64*r rows
  const int brow0 = tid >> 2;                     // + 64*r rows (r<2)

  auto issue_tile = [&](int buf, int k0) {
    #pragma unroll
    for (int r = 0; r < 4; ++r)
      async_copy16(A + (size_t)(m0 + arow0 + 64 * r) * lda + k0 + acj * 8,
                   &As[buf][arow0 + 64 * r][acj * 8]);
    #pragma unroll
    for (int r = 0; r < 2; ++r)
      async_copy16(Bt + (size_t)(n0 + brow0 + 64 * r) * ldb + k0 + acj * 8,
                   &Bs[buf][brow0 + 64 * r][acj * 8]);
  };

  union Frag { v16bf v; unsigned u[8]; };
  v8f acc[4][4] = {};

  const int kTiles = Kd / BK;
  issue_tile(0, 0);

  for (int kt = 0; kt < kTiles; ++kt) {
    const int cur = kt & 1;
    if (kt + 1 < kTiles) {
      issue_tile(cur ^ 1, (kt + 1) * BK);   // prefetch next tile
      wait_async6();                         // only the 6 new ops may be pending
    } else {
      wait_async0();
    }
    __syncthreads();

    // ---- gather WMMA fragments, contiguous k = 16*half + s in A and B ----
    Frag af[4], bf[4];
    #pragma unroll
    for (int mt = 0; mt < 4; ++mt) {
      int row = waveM * 64 + mt * 16 + l16;
      #pragma unroll
      for (int j = 0; j < 8; ++j) {
        int kk = 2 * j + half * 16;                        // A 16x32 bf16 layout
        af[mt].u[j] = *(const unsigned*)&As[cur][row][kk];
      }
    }
    #pragma unroll
    for (int nt = 0; nt < 4; ++nt) {
      int n = waveN * 64 + nt * 16 + l16;
      #pragma unroll
      for (int j = 0; j < 8; ++j) {
        int kk = 2 * j + half * 16;                        // B 32x16 bf16 layout
        bf[nt].u[j] = *(const unsigned*)&Bs[cur][n][kk];
      }
    }

    #pragma unroll
    for (int mt = 0; mt < 4; ++mt)
      #pragma unroll
      for (int nt = 0; nt < 4; ++nt)
        acc[mt][nt] = __builtin_amdgcn_wmma_f32_16x16x32_bf16(
            false, af[mt].v, false, bf[nt].v,
            (short)0, acc[mt][nt], false, false);

    __syncthreads();   // protect buffer re-filled by next iteration's prefetch
  }

  // ---- epilogue: C/D layout VGPR r -> row r + 8*half, lane -> col ----
  #pragma unroll
  for (int mt = 0; mt < 4; ++mt)
    #pragma unroll
    for (int nt = 0; nt < 4; ++nt) {
      int col  = n0 + waveN * 64 + nt * 16 + l16;
      int rowb = m0 + waveM * 64 + mt * 16 + 8 * half;
      float* cp = C + (size_t)rowb * ldc + col;
      #pragma unroll
      for (int r = 0; r < 8; ++r)
        cp[(size_t)r * ldc] = acc[mt][nt][r];
    }
}

// ---------------------------------------------------------------------------
// Causal softmax over scores row; writes attention probs as bf16
// ---------------------------------------------------------------------------
__global__ __launch_bounds__(256)
void softmax_causal_kernel(const float* __restrict__ scores,
                           unsigned short* __restrict__ P, float scale) {
  int row = blockIdx.x;                 // b*T + q
  int q = row % T_;
  const float* s = scores + (size_t)row * T_;
  unsigned short* p = P + (size_t)row * T_;
  __shared__ float red[256];

  float m = -3.4e38f;
  for (int k = threadIdx.x; k <= q; k += 256) m = fmaxf(m, s[k] * scale);
  red[threadIdx.x] = m; __syncthreads();
  for (int off = 128; off > 0; off >>= 1) {
    if (threadIdx.x < off)
      red[threadIdx.x] = fmaxf(red[threadIdx.x], red[threadIdx.x + off]);
    __syncthreads();
  }
  m = red[0]; __syncthreads();

  float sum = 0.f;
  for (int k = threadIdx.x; k <= q; k += 256) sum += __expf(s[k] * scale - m);
  red[threadIdx.x] = sum; __syncthreads();
  for (int off = 128; off > 0; off >>= 1) {
    if (threadIdx.x < off) red[threadIdx.x] += red[threadIdx.x + off];
    __syncthreads();
  }
  float inv = 1.f / red[0];

  for (int k = threadIdx.x; k < T_; k += 256)
    p[k] = (k <= q) ? f2bf(__expf(s[k] * scale - m) * inv) : (unsigned short)0;
}

// ---------------------------------------------------------------------------
// NLL loss: per-row log-softmax, mean of -logp[target]
// ---------------------------------------------------------------------------
__global__ void zero_loss_kernel(float* loss) { *loss = 0.f; }

__global__ __launch_bounds__(256)
void loss_kernel(const float* __restrict__ logits, const int* __restrict__ targets,
                 float* __restrict__ loss) {
  int row = blockIdx.x;                 // 0 .. B*T-1
  const float* l = logits + (size_t)row * V_;
  __shared__ float red[256];

  float m = -3.4e38f;
  for (int v = threadIdx.x; v < V_; v += 256) m = fmaxf(m, l[v]);
  red[threadIdx.x] = m; __syncthreads();
  for (int off = 128; off > 0; off >>= 1) {
    if (threadIdx.x < off)
      red[threadIdx.x] = fmaxf(red[threadIdx.x], red[threadIdx.x + off]);
    __syncthreads();
  }
  m = red[0]; __syncthreads();

  float sum = 0.f;
  for (int v = threadIdx.x; v < V_; v += 256) sum += __expf(l[v] - m);
  red[threadIdx.x] = sum; __syncthreads();
  for (int off = 128; off > 0; off >>= 1) {
    if (threadIdx.x < off) red[threadIdx.x] += red[threadIdx.x + off];
    __syncthreads();
  }

  if (threadIdx.x == 0) {
    float logp = l[targets[row]] - m - __logf(red[0]);
    atomicAdd(loss, -logp / (float)(B_ * T_));
  }
}

// ---------------------------------------------------------------------------
extern "C" void kernel_launch(void* const* d_in, const int* in_sizes, int n_in,
                              void* d_out, int out_size, void* d_ws, size_t ws_size,
                              hipStream_t stream) {
  const int*   inputs  = (const int*)d_in[0];
  const int*   targets = (const int*)d_in[1];
  const float* embed   = (const float*)d_in[2];
  const float* pos     = (const float*)d_in[3];
  const float* Wk      = (const float*)d_in[4];
  const float* Wq      = (const float*)d_in[5];
  // d_in[6] = Wv: intentionally unused (reference uses Wq for values)
  const float* Wlm     = (const float*)d_in[7];

  float* logits = (float*)d_out;                         // (B*T) x V
  float* loss   = logits + (size_t)out_size - 1;         // last element

  char* ws = (char*)d_ws;
  size_t off = 0;
  auto alloc = [&](size_t bytes) -> void* {
    void* p = ws + off;
    off = (off + bytes + 255) & ~(size_t)255;
    return p;
  };
  unsigned short* xbf   = (unsigned short*)alloc((size_t)B_ * T_ * D_ * 2);
  unsigned short* Wqt   = (unsigned short*)alloc((size_t)D_ * D_ * 2);   // D x D (transposed)
  unsigned short* Wkt   = (unsigned short*)alloc((size_t)D_ * D_ * 2);
  unsigned short* Wlmt  = (unsigned short*)alloc((size_t)D_ * V_ * 2);   // V x D (transposed)
  float*          Qf    = (float*)alloc((size_t)B_ * T_ * D_ * 4);
  unsigned short* Qb    = (unsigned short*)alloc((size_t)B_ * T_ * D_ * 2); // (b,T,D)
  unsigned short* Qt    = (unsigned short*)alloc((size_t)B_ * T_ * D_ * 2); // (b,D,T)
  float*          Kf    = (float*)alloc((size_t)B_ * T_ * D_ * 4);
  unsigned short* Kb    = (unsigned short*)alloc((size_t)B_ * T_ * D_ * 2);
  float*          S     = (float*)alloc((size_t)B_ * T_ * T_ * 4);
  unsigned short* P     = (unsigned short*)alloc((size_t)B_ * T_ * T_ * 2);
  float*          Of    = (float*)alloc((size_t)B_ * T_ * D_ * 4);
  unsigned short* Ob    = (unsigned short*)alloc((size_t)B_ * T_ * D_ * 2);

  const dim3 blk(256);

  // 1) x = embed[inputs] + pos  (bf16)
  embed_kernel<<<B_ * T_, blk, 0, stream>>>(inputs, embed, pos, xbf);

  // 2) transposed bf16 weights (so every GEMM is TN / K-contiguous)
  transpose_f32_to_bf16_kernel<<<dim3(D_ / 32, D_ / 32, 1), blk, 0, stream>>>(
      Wq, Wqt, D_, D_, 0, 0);
  transpose_f32_to_bf16_kernel<<<dim3(D_ / 32, D_ / 32, 1), blk, 0, stream>>>(
      Wk, Wkt, D_, D_, 0, 0);
  transpose_f32_to_bf16_kernel<<<dim3(V_ / 32, D_ / 32, 1), blk, 0, stream>>>(
      Wlm, Wlmt, D_, V_, 0, 0);

  // 3) Q = x @ Wq ; K = x @ Wk   (M=4096, N=1024, K=1024)
  gemm_bf16_tn_kernel<<<dim3(D_ / 128, (B_ * T_) / 256, 1), blk, 0, stream>>>(
      xbf, Wqt, Qf, D_, D_, D_, D_, 0, 0, 0);
  f32_to_bf16_kernel<<<(unsigned)(((long)B_ * T_ * D_ + 255) / 256), blk, 0, stream>>>(
      Qf, Qb, (long)B_ * T_ * D_);
  // Qt = per-batch transpose of Q (serves as V^T for the PV GEMM)
  transpose_f32_to_bf16_kernel<<<dim3(D_ / 32, T_ / 32, B_), blk, 0, stream>>>(
      Qf, Qt, T_, D_, (long)T_ * D_, (long)T_ * D_);

  gemm_bf16_tn_kernel<<<dim3(D_ / 128, (B_ * T_) / 256, 1), blk, 0, stream>>>(
      xbf, Wkt, Kf, D_, D_, D_, D_, 0, 0, 0);
  f32_to_bf16_kernel<<<(unsigned)(((long)B_ * T_ * D_ + 255) / 256), blk, 0, stream>>>(
      Kf, Kb, (long)B_ * T_ * D_);

  // 4) scores = Q @ K^T per batch (M=N=2048, K=1024); K is already NxK form
  gemm_bf16_tn_kernel<<<dim3(T_ / 128, T_ / 256, B_), blk, 0, stream>>>(
      Qb, Kb, S, D_, D_, D_, T_,
      (long)T_ * D_, (long)T_ * D_, (long)T_ * T_);

  // 5) causal softmax, scale = D^-0.5 = 1/32
  softmax_causal_kernel<<<B_ * T_, blk, 0, stream>>>(S, P, 0.03125f);

  // 6) outs = P @ V, V == Q (reference bug); B operand = Qt (D x T per batch)
  gemm_bf16_tn_kernel<<<dim3(D_ / 128, T_ / 256, B_), blk, 0, stream>>>(
      P, Qt, Of, T_, T_, T_, D_,
      (long)T_ * T_, (long)T_ * D_, (long)T_ * D_);
  f32_to_bf16_kernel<<<(unsigned)(((long)B_ * T_ * D_ + 255) / 256), blk, 0, stream>>>(
      Of, Ob, (long)B_ * T_ * D_);

  // 7) logits = outs @ W_lm  (M=4096, N=32000, K=1024) -> d_out
  gemm_bf16_tn_kernel<<<dim3(V_ / 128, (B_ * T_) / 256, 1), blk, 0, stream>>>(
      Ob, Wlmt, logits, D_, D_, D_, V_, 0, 0, 0);

  // 8) loss
  zero_loss_kernel<<<1, 1, 0, stream>>>(loss);
  loss_kernel<<<B_ * T_, blk, 0, stream>>>(logits, targets, loss);
}
